// InteractionLayer_53025666236778
// MI455X (gfx1250) — compile-verified, matrix operand order checked by reading, not run on hardware
//
#include <hip/hip_runtime.h>
#include <hip/hip_bf16.h>

#define D        128
#define NNODES   40000
#define NEDGES   640000

typedef __attribute__((ext_vector_type(16))) __bf16 v16bf;
typedef __attribute__((ext_vector_type(8)))  float  v8f;

// ---- generic 16-value bf16 fragment load (A from LDS, B from global bf16) --
// Lane L holds (row/col) = L&15, k-halves selected by L>>4:
// k in [kh, kh+8) and [kh+16, kh+24), kh = (L>>4)*8.  Two 16-byte loads.
__device__ __forceinline__ v16bf load_frag16(const __bf16* p) {
    union { uint4 u[2]; v16bf v; } t;
    t.u[0] = *(const uint4*)(p);
    t.u[1] = *(const uint4*)(p + 16);
    return t.v;
}

// pack 4 fp32 -> 4 bf16 (8 bytes) for LDS staging stores
union bf16x4 { uint2 u; __bf16 h[4]; };
__device__ __forceinline__ bf16x4 cvt4(float4 v) {
    bf16x4 r;
    r.h[0] = (__bf16)v.x; r.h[1] = (__bf16)v.y;
    r.h[2] = (__bf16)v.z; r.h[3] = (__bf16)v.w;
    return r;
}

// ---- zero-init for agg_sum + degree (workspace) -----------------------------
__global__ void ws_zero_kernel(float* __restrict__ p, int n) {
    for (int i = blockIdx.x * blockDim.x + threadIdx.x; i < n; i += gridDim.x * blockDim.x)
        p[i] = 0.0f;
}

// ---- one-shot fp32 -> bf16 weight conversion --------------------------------
__global__ void wcvt_kernel(const float* __restrict__ We, const float* __restrict__ Wn,
                            __bf16* __restrict__ Web, __bf16* __restrict__ Wnb) {
    const int i = blockIdx.x * blockDim.x + threadIdx.x;
    if (i < D * 3 * D) Web[i] = (__bf16)We[i];
    if (i < D * 2 * D) Wnb[i] = (__bf16)Wn[i];
}

// ---- edge update: e = [nf[src]|nf[dst]|ef] @ We^T + be, fused scatter-add ---
// 64 edges / block (4 M-tiles), 4 waves; wave w owns N-tiles {2w, 2w+1} for all
// 4 M-tiles -> 8 accumulators, B fragments amortized over 4 A fragments.
#define KE      (3 * D)     // 384
#define LDSA_E  392         // padded LDS row stride (halves): 784B rows, bank stride 4

__global__ __launch_bounds__(128) void edge_kernel(
    const float*  __restrict__ nf, const float* __restrict__ ef,
    const int*    __restrict__ src, const int* __restrict__ dst,
    const __bf16* __restrict__ Web, const float* __restrict__ be,
    float* __restrict__ e_out, float* __restrict__ agg, float* __restrict__ deg)
{
    __shared__ __align__(16) __bf16 sA[64 * LDSA_E];
    __shared__ int sSrc[64];
    __shared__ int sDst[64];

    const int tid = threadIdx.x;
    const int e0  = blockIdx.x * 64;

    if (tid < 64) { sSrc[tid] = src[e0 + tid]; sDst[tid] = dst[e0 + tid]; }
    __syncthreads();

    // Stage x_e tile [64 x 384] into LDS as bf16, float4 granularity.
    // 64*96 float4 elements; consecutive threads -> consecutive 16B -> coalesced.
    for (int idx = tid; idx < 64 * (KE / 4); idx += 128) {
        const int row = idx / (KE / 4);
        const int k4  = idx - row * (KE / 4);     // float4 index within row
        const int k   = k4 * 4;
        float4 v;
        if (k < D)            v = *(const float4*)&nf[sSrc[row] * D + k];
        else if (k < 2 * D)   v = *(const float4*)&nf[sDst[row] * D + (k - D)];
        else                  v = *(const float4*)&ef[(e0 + row) * D + (k - 2 * D)];
        *(uint2*)&sA[row * LDSA_E + k] = cvt4(v).u;
    }
    __syncthreads();

    // Degree accumulation (one lane per edge of this tile).
    if (tid < 64) atomicAdd(&deg[sDst[tid]], 1.0f);

    const int wave  = tid >> 5;
    const int lane  = tid & 31;
    const int mrow  = lane & 15;          // A row within a 16-row M-tile
    const int ncol  = lane & 15;          // C/D column within a 16-col N-tile
    const int khalf = (lane >> 4) * 8;    // per-lane k-half select
    const int nt0   = wave * 2;
    const int nt1   = wave * 2 + 1;

    const __bf16* W0 = Web + (nt0 * 16 + ncol) * KE;
    const __bf16* W1 = Web + (nt1 * 16 + ncol) * KE;

    v8f acc[4][2];
    #pragma unroll
    for (int mt = 0; mt < 4; ++mt) { acc[mt][0] = (v8f){}; acc[mt][1] = (v8f){}; }

    #pragma unroll
    for (int kc = 0; kc < KE / 32; ++kc) {
        const int kb = kc * 32 + khalf;
        __builtin_prefetch(W0 + kb + 64, 0, 3);
        const v16bf b0 = load_frag16(W0 + kb);
        const v16bf b1 = load_frag16(W1 + kb);
        #pragma unroll
        for (int mt = 0; mt < 4; ++mt) {
            const v16bf a = load_frag16(sA + (mt * 16 + mrow) * LDSA_E + kb);
            acc[mt][0] = __builtin_amdgcn_wmma_f32_16x16x32_bf16(
                false, a, false, b0, (short)0, acc[mt][0], false, false);
            acc[mt][1] = __builtin_amdgcn_wmma_f32_16x16x32_bf16(
                false, a, false, b1, (short)0, acc[mt][1], false, false);
        }
    }

    const float bias0 = be[nt0 * 16 + ncol];
    const float bias1 = be[nt1 * 16 + ncol];

    #pragma unroll
    for (int mt = 0; mt < 4; ++mt) {
        #pragma unroll
        for (int r = 0; r < 8; ++r) {
            const int m   = mt * 16 + r + ((lane >> 4) * 8);  // C/D row mapping
            const int eid = e0 + m;
            const int dn  = sDst[m];
            const float v0 = acc[mt][0][r] + bias0;
            const float v1 = acc[mt][1][r] + bias1;
            // e is written once and never re-read on-chip: stream it out NT.
            __builtin_nontemporal_store(v0, &e_out[eid * D + nt0 * 16 + ncol]);
            __builtin_nontemporal_store(v1, &e_out[eid * D + nt1 * 16 + ncol]);
            // Scatter-add into agg_sum (20.5 MB -> L2-resident atomics).
            atomicAdd(&agg[dn * D + nt0 * 16 + ncol], v0);
            atomicAdd(&agg[dn * D + nt1 * 16 + ncol], v1);
        }
    }
}

// ---- node update: n = [nf | agg_sum/deg] @ Wn^T + bn ------------------------
#define KN      (2 * D)     // 256
#define LDSA_N  264         // padded LDS row stride (halves)

__global__ __launch_bounds__(128) void node_kernel(
    const float*  __restrict__ nf, const __bf16* __restrict__ Wnb,
    const float*  __restrict__ bn, const float* __restrict__ agg,
    const float*  __restrict__ deg, float* __restrict__ n_out)
{
    __shared__ __align__(16) __bf16 sA[64 * LDSA_N];

    const int tid = threadIdx.x;
    const int n0  = blockIdx.x * 64;

    // Stage x_n tile [64 x 256]; fuse the segment-mean (/deg, 0 if no edges).
    for (int idx = tid; idx < 64 * (KN / 4); idx += 128) {
        const int row  = idx >> 6;                 // /64 float4 per row
        const int k4   = idx & 63;
        const int k    = k4 * 4;
        const int node = n0 + row;
        float4 v;
        if (k < D) {
            v = *(const float4*)&nf[node * D + k];
        } else {
            const float dg  = deg[node];
            const float inv = (dg > 0.0f) ? 1.0f / dg : 0.0f;
            float4 s = *(const float4*)&agg[node * D + (k - D)];
            v = make_float4(s.x * inv, s.y * inv, s.z * inv, s.w * inv);
        }
        *(uint2*)&sA[row * LDSA_N + k] = cvt4(v).u;
    }
    __syncthreads();

    const int wave  = tid >> 5;
    const int lane  = tid & 31;
    const int mrow  = lane & 15;
    const int ncol  = lane & 15;
    const int khalf = (lane >> 4) * 8;
    const int nt0   = wave * 2;
    const int nt1   = wave * 2 + 1;

    const __bf16* W0 = Wnb + (nt0 * 16 + ncol) * KN;
    const __bf16* W1 = Wnb + (nt1 * 16 + ncol) * KN;

    v8f acc[4][2];
    #pragma unroll
    for (int mt = 0; mt < 4; ++mt) { acc[mt][0] = (v8f){}; acc[mt][1] = (v8f){}; }

    #pragma unroll
    for (int kc = 0; kc < KN / 32; ++kc) {
        const int kb = kc * 32 + khalf;
        const v16bf b0 = load_frag16(W0 + kb);
        const v16bf b1 = load_frag16(W1 + kb);
        #pragma unroll
        for (int mt = 0; mt < 4; ++mt) {
            const v16bf a = load_frag16(sA + (mt * 16 + mrow) * LDSA_N + kb);
            acc[mt][0] = __builtin_amdgcn_wmma_f32_16x16x32_bf16(
                false, a, false, b0, (short)0, acc[mt][0], false, false);
            acc[mt][1] = __builtin_amdgcn_wmma_f32_16x16x32_bf16(
                false, a, false, b1, (short)0, acc[mt][1], false, false);
        }
    }

    const float bias0 = bn[nt0 * 16 + ncol];
    const float bias1 = bn[nt1 * 16 + ncol];

    #pragma unroll
    for (int mt = 0; mt < 4; ++mt) {
        #pragma unroll
        for (int r = 0; r < 8; ++r) {
            const int m    = mt * 16 + r + ((lane >> 4) * 8);
            const int node = n0 + m;
            n_out[node * D + nt0 * 16 + ncol] = acc[mt][0][r] + bias0;
            n_out[node * D + nt1 * 16 + ncol] = acc[mt][1][r] + bias1;
        }
    }
}

// ---- launcher ---------------------------------------------------------------
extern "C" void kernel_launch(void* const* d_in, const int* in_sizes, int n_in,
                              void* d_out, int out_size, void* d_ws, size_t ws_size,
                              hipStream_t stream) {
    const float* nf  = (const float*)d_in[0];   // [N, D]
    const float* ef  = (const float*)d_in[1];   // [E, D]
    const int*   src = (const int*)  d_in[2];   // [E]
    const int*   dst = (const int*)  d_in[3];   // [E]
    const float* We  = (const float*)d_in[4];   // [D, 3D]
    const float* be  = (const float*)d_in[5];   // [D]
    const float* Wn  = (const float*)d_in[6];   // [D, 2D]
    const float* bn  = (const float*)d_in[7];   // [D]

    float* n_out = (float*)d_out;                       // [N, D] first
    float* e_out = n_out + (size_t)NNODES * D;          // [E, D] second

    // workspace layout: agg_sum | deg | We(bf16) | Wn(bf16)
    float*  agg = (float*)d_ws;                         // [N, D]
    float*  deg = agg + (size_t)NNODES * D;             // [N]
    __bf16* Web = (__bf16*)(deg + NNODES);              // [D, 3D] bf16
    __bf16* Wnb = Web + (size_t)D * 3 * D;              // [D, 2D] bf16

    const int zn = NNODES * D + NNODES;
    ws_zero_kernel<<<512, 256, 0, stream>>>(agg, zn);
    wcvt_kernel<<<(D * 3 * D + 255) / 256, 256, 0, stream>>>(We, Wn, Web, Wnb);

    edge_kernel<<<NEDGES / 64, 128, 0, stream>>>(nf, ef, src, dst, Web, be,
                                                 e_out, agg, deg);

    node_kernel<<<NNODES / 64, 128, 0, stream>>>(nf, Wnb, bn, agg, deg, n_out);
}
